// PsRoiOffset_16243566314021
// MI455X (gfx1250) — matrix operand order repaired and checked by reading, not compiled, not run on hardware
//
#include <hip/hip_runtime.h>
#include <hip/hip_bf16.h>

// Problem constants (from reference)
#define KKB   7                 // PS-ROI bins per side
#define HH    200
#define WW    304
#define CC    196               // 4*K*K feature channels
#define C2    392               // 2*CC offset channels
#define NROIS 512

// GEMM tiling constants
#define KPAD   224              // per-tap channels padded to multiple of 32
#define KTOT   (9 * KPAD)       // 2016 total K (tap-major)
#define NPAD   400              // 392 -> 25 tiles of 16
#define NTILES 25
#define MPIX   (HH * WW)        // 60800 = 3800 * 16 exactly
#define MTILES (MPIX / 16)      // 3800

// Workspace layout (bytes, all 256-aligned)
#define WS_FB_OFF   0ull                       // bf16 features  [MPIX][KPAD]  = 27,238,400 B
#define WS_WB_OFF   27238400ull                // bf16 weights   [NPAD][KTOT]  =  1,612,800 B
#define WS_OMAP_OFF 28851200ull                // f32 offset_map [MPIX][CC]    = 47,667,200 B

typedef __attribute__((ext_vector_type(16))) __bf16 v16bf;
typedef __attribute__((ext_vector_type(8)))  float  v8f;

__device__ __forceinline__ unsigned short f2bf(float x) {
    // round-to-nearest-even f32 -> bf16
    unsigned int u = __float_as_uint(x);
    unsigned int r = u + 0x7FFFu + ((u >> 16) & 1u);
    return (unsigned short)(r >> 16);
}

// ---------------------------------------------------------------------------
// Pack features f32 (H,W,196) -> bf16 (H*W, 224) zero-padded K
// ---------------------------------------------------------------------------
__global__ void pack_feat_kernel(const float* __restrict__ f,
                                 unsigned short* __restrict__ Fb) {
    long long t = (long long)blockIdx.x * blockDim.x + threadIdx.x;
    if (t >= (long long)MPIX * KPAD) return;
    int c = (int)(t % KPAD);
    long long p = t / KPAD;
    float v = (c < CC) ? f[p * CC + c] : 0.0f;
    Fb[t] = f2bf(v);
}

// ---------------------------------------------------------------------------
// Pack weights f32 (3,3,196,392) -> bf16 N-major (400, 2016):
//   Wb[n][tap*KPAD + c] = W_off[tap][c][n]   (zero pad c>=196, n>=392)
// ---------------------------------------------------------------------------
__global__ void pack_w_kernel(const float* __restrict__ Wo,
                              unsigned short* __restrict__ Wb) {
    long long t = (long long)blockIdx.x * blockDim.x + threadIdx.x;
    if (t >= (long long)NPAD * KTOT) return;
    int kk = (int)(t % KTOT);
    int n  = (int)(t / KTOT);
    int tap = kk / KPAD;
    int c   = kk % KPAD;
    float v = (c < CC && n < C2) ? Wo[((long long)(tap * CC + c)) * C2 + n] : 0.0f;
    Wb[t] = f2bf(v);
}

// ---------------------------------------------------------------------------
// Stage one tap's B tile (16 N-rows x 224 K bf16 = 7168 B) into LDS using the
// CDNA5 async global->LDS path (tracked by ASYNCcnt). 448 x 16-byte chunks
// split over 256 threads.
// ---------------------------------------------------------------------------
__device__ __forceinline__ void stage_b_tap(const unsigned short* __restrict__ Wb,
                                            unsigned short* sdst,
                                            int nTile, int tap, int tid) {
#pragma unroll
    for (int i = 0; i < 2; ++i) {
        int ch = tid + i * 256;
        if (ch < 448) {
            int n  = ch / 28;            // N row 0..15
            int km = ch % 28;            // 8-element K chunk 0..27
            unsigned ldsb = (unsigned)(uintptr_t)(sdst + n * KPAD + km * 8);
            unsigned goff = (unsigned)((((nTile * 16 + n) * KTOT) +
                                        tap * KPAD + km * 8) * 2);
            asm volatile("global_load_async_to_lds_b128 %0, %1, %2"
                         :: "v"(ldsb), "v"(goff), "s"(Wb)
                         : "memory");
        }
    }
}

__device__ __forceinline__ void wait_async_zero() {
    asm volatile("s_wait_asynccnt 0x0" ::: "memory");
}

// ---------------------------------------------------------------------------
// Fused conv3x3(SAME) offset GEMM via v_wmma_f32_16x16x32_bf16 + bias +
// per-pixel deformable bilinear sampling epilogue.
// One wave per 16(pixel) x 16(offset-channel) tile; all 8 waves of a block
// share nTile, so the B tile is staged in LDS once per tap (double-buffered).
// ---------------------------------------------------------------------------
__global__ __launch_bounds__(256)
void conv_off_sample_kernel(const float* __restrict__ feat,
                            const float* __restrict__ bias,
                            const unsigned short* __restrict__ Fb,
                            const unsigned short* __restrict__ Wb,
                            float* __restrict__ omap) {
    __shared__ unsigned short sB[2][16 * KPAD];   // 2 x 7168 B double buffer

    const int lane  = threadIdx.x;                 // 0..31
    const int tid   = threadIdx.y * 32 + threadIdx.x;
    const int mTile = blockIdx.x * 8 + threadIdx.y;
    const int nTile = blockIdx.y;                  // 0..24
    const int hi    = lane >> 4;                   // lane half select
    const int rowA  = lane & 15;                   // A-matrix row (M) for this lane

    // Pixel this lane's A row corresponds to (M = rowA)
    const int pixA = mTile * 16 + rowA;            // MPIX is exact multiple of 16
    const int hA = pixA / WW, wA = pixA % WW;

    // Prologue: stage tap 0 weights
    stage_b_tap(Wb, &sB[0][0], nTile, 0, tid);
    wait_async_zero();
    __syncthreads();

    v8f acc = {};
    for (int tap = 0; tap < 9; ++tap) {
        const int buf = tap & 1;
        // Prefetch next tap's weights into the other buffer (async)
        if (tap < 8) stage_b_tap(Wb, &sB[buf ^ 1][0], nTile, tap + 1, tid);

        const int dy = tap / 3 - 1, dx = tap % 3 - 1;
        const int hh = hA + dy, ww2 = wA + dx;
        const bool inb = (hh >= 0) && (hh < HH) && (ww2 >= 0) && (ww2 < WW);
        const unsigned short* Arow =
            Fb + (size_t)(inb ? (hh * WW + ww2) : 0) * KPAD;
        const unsigned short* Bl = &sB[buf][(lane & 15) * KPAD];
        const uint4 z = make_uint4(0u, 0u, 0u, 0u);
#pragma unroll
        for (int kc = 0; kc < 7; ++kc) {
            // A 16x32, contiguous k order used here (same as B): lanes 0-15 hold K 0..15,
            // lanes 16-31 hold K 16..31  (relative to kc*32)
            const int cbase = kc * 32 + hi * 16;
            uint4 a0 = inb ? *(const uint4*)(Arow + cbase)      : z;
            uint4 a1 = inb ? *(const uint4*)(Arow + cbase + 8) : z;
            union { uint4 u[2]; v16bf v; } A;
            A.u[0] = a0; A.u[1] = a1;
            // B 32x16 from LDS: lanes 0-15 K=0..15, lanes 16-31 K=16..31
            v16bf Bf = *(const v16bf*)(Bl + kc * 32 + hi * 16);
            acc = __builtin_amdgcn_wmma_f32_16x16x32_bf16(
                false, A.v, false, Bf, (short)0, acc, false, false);
        }

        if (tap < 8) {
            wait_async_zero();   // this wave's async stores to LDS landed
            __syncthreads();     // whole block: next buffer ready, cur buffer free
        }
    }

    // Epilogue: bias add. C/D layout: VGPR r holds M = r + 8*hi, N = lane&15.
    const int colc = nTile * 16 + (lane & 15);     // offset channel in [0,400)
    const float b = (colc < C2) ? bias[colc] : 0.0f;
#pragma unroll
    for (int r = 0; r < 8; ++r) acc[r] += b;

    // Pair up (y,x) offsets: channel 2c (even lane) / 2c+1 (odd lane)
    float part[8];
#pragma unroll
    for (int r = 0; r < 8; ++r) part[r] = __shfl_xor(acc[r], 1, 32);

    if (((lane & 1) == 0) && ((colc >> 1) < CC)) {
        const int c = colc >> 1;                   // feature channel
#pragma unroll
        for (int r = 0; r < 8; ++r) {
            const int p = mTile * 16 + (r + 8 * hi);
            const int h = p / WW, w = p % WW;
            float yy = fminf(fmaxf((float)h + acc[r], 0.0f), (float)(HH - 1));
            float xx = fminf(fmaxf((float)w + part[r], 0.0f), (float)(WW - 1));
            float y0 = floorf(yy), x0 = floorf(xx);
            float wy = yy - y0, wx = xx - x0;
            int y0i = (int)y0, x0i = (int)x0;
            int y1i = min(y0i + 1, HH - 1), x1i = min(x0i + 1, WW - 1);
            float v00 = feat[((size_t)y0i * WW + x0i) * CC + c];
            float v01 = feat[((size_t)y0i * WW + x1i) * CC + c];
            float v10 = feat[((size_t)y1i * WW + x0i) * CC + c];
            float v11 = feat[((size_t)y1i * WW + x1i) * CC + c];
            float top = v00 + (v01 - v00) * wx;
            float bot = v10 + (v11 - v10) * wx;
            omap[(size_t)p * CC + c] = top + (bot - top) * wy;
        }
    }
}

// ---------------------------------------------------------------------------
// PS-ROI align: one thread per (roi, gy, gx, d); 2x2 samples, bilinear.
// Output (512, 7, 7, 4) f32.
// ---------------------------------------------------------------------------
__global__ void ps_roi_kernel(const float* __restrict__ omap,
                              const float* __restrict__ rois,
                              float* __restrict__ out) {
    int t = blockIdx.x * blockDim.x + threadIdx.x;
    if (t >= NROIS * KKB * KKB * 4) return;
    int d    = t & 3;
    int rest = t >> 2;
    int gx   = rest % KKB;
    int gy   = (rest / KKB) % KKB;
    int roi  = rest / (KKB * KKB);

    float x1 = rois[roi * 4 + 0] * 0.25f;   // / FEAT_STRIDE
    float y1 = rois[roi * 4 + 1] * 0.25f;
    float x2 = rois[roi * 4 + 2] * 0.25f;
    float y2 = rois[roi * 4 + 3] * 0.25f;
    float bh = (y2 - y1) * (1.0f / 7.0f);
    float bw = (x2 - x1) * (1.0f / 7.0f);
    int ch = (gy * KKB + gx) * 4 + d;

    float sum = 0.0f;
#pragma unroll
    for (int sy = 0; sy < 2; ++sy) {
        float sfy = ((float)sy + 0.5f) * 0.5f;
        float yy = fminf(fmaxf(y1 + ((float)gy + sfy) * bh, 0.0f), (float)(HH - 1));
        float y0 = floorf(yy);
        float wy = yy - y0;
        int y0i = (int)y0;
        int y1i = min(y0i + 1, HH - 1);
#pragma unroll
        for (int sx = 0; sx < 2; ++sx) {
            float sfx = ((float)sx + 0.5f) * 0.5f;
            float xx = fminf(fmaxf(x1 + ((float)gx + sfx) * bw, 0.0f), (float)(WW - 1));
            float x0 = floorf(xx);
            float wx = xx - x0;
            int x0i = (int)x0;
            int x1i = min(x0i + 1, WW - 1);
            float v00 = omap[((size_t)y0i * WW + x0i) * CC + ch];
            float v01 = omap[((size_t)y0i * WW + x1i) * CC + ch];
            float v10 = omap[((size_t)y1i * WW + x0i) * CC + ch];
            float v11 = omap[((size_t)y1i * WW + x1i) * CC + ch];
            float top = v00 + (v01 - v00) * wx;
            float bot = v10 + (v11 - v10) * wx;
            sum += top + (bot - top) * wy;
        }
    }
    out[t] = sum * 0.25f;
}

// ---------------------------------------------------------------------------
extern "C" void kernel_launch(void* const* d_in, const int* in_sizes, int n_in,
                              void* d_out, int out_size, void* d_ws, size_t ws_size,
                              hipStream_t stream) {
    const float* feat = (const float*)d_in[0];   // (1,200,304,196)
    const float* rois = (const float*)d_in[1];   // (512,4)
    const float* Woff = (const float*)d_in[2];   // (3,3,196,392)
    const float* boff = (const float*)d_in[3];   // (392,)

    char* ws = (char*)d_ws;
    unsigned short* Fb   = (unsigned short*)(ws + WS_FB_OFF);
    unsigned short* Wb   = (unsigned short*)(ws + WS_WB_OFF);
    float*          omap = (float*)(ws + WS_OMAP_OFF);

    {
        long long n = (long long)MPIX * KPAD;
        pack_feat_kernel<<<(unsigned)((n + 255) / 256), 256, 0, stream>>>(feat, Fb);
    }
    {
        long long n = (long long)NPAD * KTOT;
        pack_w_kernel<<<(unsigned)((n + 255) / 256), 256, 0, stream>>>(Woff, Wb);
    }
    conv_off_sample_kernel<<<dim3(MTILES / 8, NTILES), dim3(32, 8), 0, stream>>>(
        feat, boff, Fb, Wb, omap);
    ps_roi_kernel<<<(NROIS * KKB * KKB * 4 + 255) / 256, 256, 0, stream>>>(
        omap, rois, (float*)d_out);
}
